// GatedDeltaNet_16655883174525
// MI455X (gfx1250) — matrix operand order repaired and checked by reading, not compiled, not run on hardware
//
#include <hip/hip_runtime.h>
#include <hip/hip_bf16.h>
#include <math.h>

// ---------------- problem constants ----------------
#define HID 2048
#define NK  16
#define DK  128
#define NV  32
#define DV  128
#define KC  4
#define PP  2            // NV/NK
#define SV  256          // PP*DV
#define QD  2048         // NK*DK
#define VD  4096         // NV*DV
#define QKV 8192
#define BB  2
#define TT  2048
#define MROWS (BB*TT)    // 4096

typedef __bf16 bf16_t;
typedef __attribute__((ext_vector_type(16))) __bf16 bf16x16;
typedef __attribute__((ext_vector_type(8)))  float  f32x8;
typedef __attribute__((ext_vector_type(4)))  unsigned int u32x4;
typedef __attribute__((ext_vector_type(8)))  unsigned int u32x8;

union ABfrag { bf16x16 v; uint4 q[2]; unsigned short s[16]; };

__device__ __forceinline__ unsigned short f2bf(float x) {
    union { float f; unsigned u; } v; v.f = x;
    unsigned r = v.u + 0x7FFFu + ((v.u >> 16) & 1u);   // round-to-nearest-even
    return (unsigned short)(r >> 16);
}
__device__ __forceinline__ float bf2f(unsigned short b) {
    union { unsigned u; float f; } v; v.u = ((unsigned)b) << 16; return v.f;
}

// =====================================================================
// Tensor Data Mover: 2D tile (tile_k x tile_m) of 2-byte elems,
// row stride = strideElems, global tile start -> LDS byte addr.
// D# per CDNA5 ISA; 2D -> groups 0,1 only.
// =====================================================================
__device__ __forceinline__ void tdm_load_2d(unsigned lds_byte_addr,
                                            const unsigned short* gtile,
                                            unsigned tile_k, unsigned tile_m,
                                            unsigned strideElems)
{
    unsigned long long ga = (unsigned long long)(uintptr_t)gtile;
    u32x4 g0;
    g0[0] = 1u;                                   // count=1, user mode
    g0[1] = lds_byte_addr;                        // lds_addr
    g0[2] = (unsigned)(ga & 0xFFFFFFFFu);         // global_addr[31:0]
    g0[3] = (unsigned)((ga >> 32) & 0x01FFFFFFu)  // global_addr[56:32]
          | (2u << 30);                           // type = 2 ("image")
    const unsigned dim0 = 0x40000000u;            // huge: OOB never triggers
    const unsigned dim1 = 0x40000000u;
    u32x8 g1;
    g1[0] = (1u << 16);                                   // data_size=1 (2B)
    g1[1] = (dim0 & 0xFFFFu) << 16;                       // tensor_dim0[15:0]
    g1[2] = ((dim0 >> 16) & 0xFFFFu) | ((dim1 & 0xFFFFu) << 16);
    g1[3] = ((dim1 >> 16) & 0xFFFFu) | ((tile_k & 0xFFFFu) << 16); // tile_dim0
    g1[4] = (tile_m & 0xFFFFu);                           // tile_dim1 (tile_dim2=0)
    g1[5] = strideElems;                                  // tensor_dim0_stride[31:0]
    g1[6] = 0;                                            // stride hi / dim1_stride lo
    g1[7] = 0;
    asm volatile("tensor_load_to_lds %0, %1" :: "s"(g0), "s"(g1) : "memory");
}

// =====================================================================
// GEMM: C[M,N] = A[M,K] * W[N,K]^T, A/W pre-split into bf16 hi/lo planes.
// Block 256 thr = 8 waves; block tile 128x64; wave tile 32x32.
// TDM double-buffered staging; inner loop is ds_load_b128 + WMMA only.
// M mult of 128, N mult of 64, K mult of 32.
// =====================================================================
#define TDM_BUF 12288                 // ushorts per buffer (24 KB)
#define OFF_AHI 0                     // 128x32
#define OFF_ALO 4096
#define OFF_BHI 8192                  // 64x32
#define OFF_BLO 10240

__global__ __launch_bounds__(256) void gemm_tdm(
    const unsigned short* __restrict__ Ahi, const unsigned short* __restrict__ Alo,
    const unsigned short* __restrict__ Whi, const unsigned short* __restrict__ Wlo,
    float* __restrict__ C, int M, int N, int K)
{
    __shared__ unsigned short smem[2 * TDM_BUF];

    const int tid   = threadIdx.x;
    const int mBase = blockIdx.y * 128;
    const int nBase = blockIdx.x * 64;
    const int wave  = tid >> 5;
    const int lane  = tid & 31;
    const int mw    = wave & 3;        // 4 M groups of 32 rows
    const int nw    = wave >> 2;       // 2 N groups of 32 cols
    const int laneM = lane & 15;
    const int hiLn  = lane >> 4;

    const unsigned ldsBase = (unsigned)(uintptr_t)&smem[0];
    const bool w0 = (tid < 32);

    f32x8 acc00 = {}, acc01 = {}, acc10 = {}, acc11 = {};

    const int nIter = K >> 5;

    // prologue: stage tile 0 into buffer 0
    if (w0) {
        const unsigned b = ldsBase;
        tdm_load_2d(b + 2 * OFF_AHI, Ahi + (size_t)mBase * K, 32, 128, (unsigned)K);
        tdm_load_2d(b + 2 * OFF_ALO, Alo + (size_t)mBase * K, 32, 128, (unsigned)K);
        tdm_load_2d(b + 2 * OFF_BHI, Whi + (size_t)nBase * K, 32, 64,  (unsigned)K);
        tdm_load_2d(b + 2 * OFF_BLO, Wlo + (size_t)nBase * K, 32, 64,  (unsigned)K);
    }

    for (int it = 0; it < nIter; ++it) {
        const int cur = it & 1;
        if (w0) {
            if (it + 1 < nIter) {     // prefetch next tile into other buffer
                const unsigned b = ldsBase + (unsigned)((cur ^ 1) * TDM_BUF * 2);
                const int k1 = (it + 1) << 5;
                tdm_load_2d(b + 2 * OFF_AHI, Ahi + (size_t)mBase * K + k1, 32, 128, (unsigned)K);
                tdm_load_2d(b + 2 * OFF_ALO, Alo + (size_t)mBase * K + k1, 32, 128, (unsigned)K);
                tdm_load_2d(b + 2 * OFF_BHI, Whi + (size_t)nBase * K + k1, 32, 64,  (unsigned)K);
                tdm_load_2d(b + 2 * OFF_BLO, Wlo + (size_t)nBase * K + k1, 32, 64,  (unsigned)K);
                __builtin_amdgcn_s_wait_tensorcnt((short)4);  // oldest batch done
            } else {
                __builtin_amdgcn_s_wait_tensorcnt((short)0);
            }
        }
        __syncthreads();   // buffer `cur` visible to all waves

        const unsigned short* buf = &smem[cur * TDM_BUF];

        // A fragments: rows mw*32 + {0,16} + laneM  (K = (e/8)*16 + hiLn*8 + e%8)
        ABfrag aHi0, aHi1, aLo0, aLo1;
        {
            const uint4* p0 = (const uint4*)&buf[OFF_AHI + (mw * 32 + laneM) * 32];
            aHi0.q[0] = p0[hiLn]; aHi0.q[1] = p0[2 + hiLn];
            const uint4* p1 = (const uint4*)&buf[OFF_AHI + (mw * 32 + 16 + laneM) * 32];
            aHi1.q[0] = p1[hiLn]; aHi1.q[1] = p1[2 + hiLn];
            const uint4* p2 = (const uint4*)&buf[OFF_ALO + (mw * 32 + laneM) * 32];
            aLo0.q[0] = p2[hiLn]; aLo0.q[1] = p2[2 + hiLn];
            const uint4* p3 = (const uint4*)&buf[OFF_ALO + (mw * 32 + 16 + laneM) * 32];
            aLo1.q[0] = p3[hiLn]; aLo1.q[1] = p3[2 + hiLn];
        }
        // B fragments: rows nw*32 + {0,16} + laneM (K = (e/8)*16 + hiLn*8 + e%8)
        ABfrag bHi0, bHi1, bLo0, bLo1;
        {
            const uint4* p0 = (const uint4*)&buf[OFF_BHI + (nw * 32 + laneM) * 32];
            bHi0.q[0] = p0[hiLn]; bHi0.q[1] = p0[2 + hiLn];
            const uint4* p1 = (const uint4*)&buf[OFF_BHI + (nw * 32 + 16 + laneM) * 32];
            bHi1.q[0] = p1[hiLn]; bHi1.q[1] = p1[2 + hiLn];
            const uint4* p2 = (const uint4*)&buf[OFF_BLO + (nw * 32 + laneM) * 32];
            bLo0.q[0] = p2[hiLn]; bLo0.q[1] = p2[2 + hiLn];
            const uint4* p3 = (const uint4*)&buf[OFF_BLO + (nw * 32 + 16 + laneM) * 32];
            bLo1.q[0] = p3[hiLn]; bLo1.q[1] = p3[2 + hiLn];
        }

        // 12 WMMAs: hi*hi + hi*lo + lo*hi per (ms,ns)
        acc00 = __builtin_amdgcn_wmma_f32_16x16x32_bf16(false, aHi0.v, false, bHi0.v, (short)0, acc00, false, false);
        acc00 = __builtin_amdgcn_wmma_f32_16x16x32_bf16(false, aHi0.v, false, bLo0.v, (short)0, acc00, false, false);
        acc00 = __builtin_amdgcn_wmma_f32_16x16x32_bf16(false, aLo0.v, false, bHi0.v, (short)0, acc00, false, false);

        acc01 = __builtin_amdgcn_wmma_f32_16x16x32_bf16(false, aHi0.v, false, bHi1.v, (short)0, acc01, false, false);
        acc01 = __builtin_amdgcn_wmma_f32_16x16x32_bf16(false, aHi0.v, false, bLo1.v, (short)0, acc01, false, false);
        acc01 = __builtin_amdgcn_wmma_f32_16x16x32_bf16(false, aLo0.v, false, bHi1.v, (short)0, acc01, false, false);

        acc10 = __builtin_amdgcn_wmma_f32_16x16x32_bf16(false, aHi1.v, false, bHi0.v, (short)0, acc10, false, false);
        acc10 = __builtin_amdgcn_wmma_f32_16x16x32_bf16(false, aHi1.v, false, bLo0.v, (short)0, acc10, false, false);
        acc10 = __builtin_amdgcn_wmma_f32_16x16x32_bf16(false, aLo1.v, false, bHi0.v, (short)0, acc10, false, false);

        acc11 = __builtin_amdgcn_wmma_f32_16x16x32_bf16(false, aHi1.v, false, bHi1.v, (short)0, acc11, false, false);
        acc11 = __builtin_amdgcn_wmma_f32_16x16x32_bf16(false, aHi1.v, false, bLo1.v, (short)0, acc11, false, false);
        acc11 = __builtin_amdgcn_wmma_f32_16x16x32_bf16(false, aLo1.v, false, bHi1.v, (short)0, acc11, false, false);

        __syncthreads();   // license next iteration's TDM overwrite
    }

    // C/D layout: VGPR r -> M = r + 8*hiLn, N = laneM
    #pragma unroll
    for (int r = 0; r < 8; ++r) {
        const int mr0 = mBase + mw * 32 + hiLn * 8 + r;
        const int mr1 = mr0 + 16;
        const int nc0 = nBase + nw * 32 + laneM;
        C[(size_t)mr0 * N + nc0]      = acc00[r];
        C[(size_t)mr0 * N + nc0 + 16] = acc01[r];
        C[(size_t)mr1 * N + nc0]      = acc10[r];
        C[(size_t)mr1 * N + nc0 + 16] = acc11[r];
    }
}

// =====================================================================
// f32 -> bf16 hi/lo split (4 elems per thread)
// =====================================================================
__global__ __launch_bounds__(256) void cvt_kernel(
    const float* __restrict__ src, unsigned short* __restrict__ hi,
    unsigned short* __restrict__ lo, long long n4)
{
    long long i = (long long)blockIdx.x * 256 + threadIdx.x;
    if (i >= n4) return;
    float4 v = ((const float4*)src)[i];
    ushort4 h, l;
    h.x = f2bf(v.x); l.x = f2bf(v.x - bf2f(h.x));
    h.y = f2bf(v.y); l.y = f2bf(v.y - bf2f(h.y));
    h.z = f2bf(v.z); l.z = f2bf(v.z - bf2f(h.z));
    h.w = f2bf(v.w); l.w = f2bf(v.w - bf2f(h.w));
    ((ushort4*)hi)[i] = h;
    ((ushort4*)lo)[i] = l;
}

// =====================================================================
// beta / dt / decay:  one block per (b,t) row, 64 threads
// =====================================================================
__global__ __launch_bounds__(64) void betadt_kernel(
    const float* __restrict__ x, const float* __restrict__ Wb,
    const float* __restrict__ Wa, const float* __restrict__ dt_bias,
    const float* __restrict__ A_log, float* __restrict__ beta,
    float* __restrict__ decay)
{
    __shared__ float xs[HID];
    const int bt  = blockIdx.x;
    const int tid = threadIdx.x;
    const float* xr = &x[(size_t)bt * HID];
    for (int i = tid; i < HID; i += 64) xs[i] = xr[i];
    __syncthreads();

    const int h = tid & 31;
    const float* wr = (tid < 32) ? &Wb[(size_t)h * HID] : &Wa[(size_t)h * HID];
    const float4* x4 = (const float4*)xs;
    const float4* w4 = (const float4*)wr;
    float s = 0.f;
    for (int i = 0; i < HID / 4; ++i) {
        float4 a = x4[i], w = w4[i];
        s += a.x * w.x + a.y * w.y + a.z * w.z + a.w * w.w;
    }
    if (tid < 32) {
        beta[(size_t)bt * NV + h] = 1.f / (1.f + expf(-s));
    } else {
        float d  = s + dt_bias[h];
        float sp = (d > 20.f) ? d : log1pf(expf(d));
        decay[(size_t)bt * NV + h] = expf(-sp * expf(A_log[h]));
    }
}

// =====================================================================
// causal depthwise conv1d (KC=4, zero left pad)
// =====================================================================
__global__ __launch_bounds__(256) void conv_kernel(
    const float* __restrict__ qkv, const float* __restrict__ conv_w,
    float* __restrict__ qkv_c)
{
    size_t idx = (size_t)blockIdx.x * 256 + threadIdx.x;
    const size_t total = (size_t)BB * TT * QKV;
    if (idx >= total) return;
    const int c = (int)(idx % QKV);
    const size_t bt = idx / QKV;
    const int t = (int)(bt % TT);
    const size_t bbase = (bt / TT) * (size_t)TT * QKV;
    float acc = 0.f;
    #pragma unroll
    for (int jj = 0; jj < KC; ++jj) {
        int ts = t - (KC - 1) + jj;
        if (ts >= 0) acc += qkv[bbase + (size_t)ts * QKV + c] * conv_w[c * KC + jj];
    }
    qkv_c[idx] = acc;
}

__global__ __launch_bounds__(256) void convbuf_kernel(
    const float* __restrict__ qkv, float* __restrict__ cbuf)
{
    int idx = blockIdx.x * 256 + threadIdx.x;           // over B*QKV*(KC-1)
    if (idx >= BB * QKV * (KC - 1)) return;
    const int j = idx % (KC - 1);
    const int c = (idx / (KC - 1)) % QKV;
    const int b = idx / ((KC - 1) * QKV);
    cbuf[idx] = qkv[((size_t)b * TT + (TT - (KC - 1) + j)) * QKV + c];
}

// =====================================================================
// gated delta-rule scan: 1 block per (b, key-head); 256 thr each own a
// state column S[0..127][j] in VGPRs. L2-norm of q/k, RMSNorm, silu(z)
// gating; writes gated output directly as bf16 hi/lo for final GEMM.
// =====================================================================
__global__ __launch_bounds__(256) void scan_kernel(
    const float* __restrict__ qkv_c, const float* __restrict__ z,
    const float* __restrict__ beta, const float* __restrict__ decay,
    const float* __restrict__ norm_w, unsigned short* __restrict__ og_hi,
    unsigned short* __restrict__ og_lo, float* __restrict__ S_out)
{
    __shared__ float kb[DK];
    __shared__ float qb[DK];
    __shared__ float o2[256];
    __shared__ float scal[4];

    const int bh = blockIdx.x;          // b*NK + h
    const int h  = bh & (NK - 1);
    const int b  = bh >> 4;
    const int j  = threadIdx.x;         // 0..255 : p = j/128, jv = j%128
    const int p  = j >> 7;
    const int jv = j & 127;

    float S[DK];
    #pragma unroll
    for (int i = 0; i < DK; ++i) S[i] = 0.f;
    const float nw = norm_w[jv];

    for (int t = 0; t < TT; ++t) {
        const size_t base = ((size_t)b * TT + t) * QKV;
        if (j < DK) kb[j]        = qkv_c[base + QD + h * DK + j];
        else        qb[j - DK]   = qkv_c[base + h * DK + (j - DK)];
        const float vv = qkv_c[base + 2 * QD + h * SV + j];
        const float zv = z[((size_t)b * TT + t) * VD + h * SV + j];
        const int   bi = (b * TT + t) * NV + h * PP + p;
        const float bet = beta[bi];
        const float dec = decay[bi];
        __syncthreads();

        if (j < 32) {
            float s = kb[j] * kb[j] + kb[j + 32] * kb[j + 32] +
                      kb[j + 64] * kb[j + 64] + kb[j + 96] * kb[j + 96];
            for (int off = 16; off; off >>= 1) s += __shfl_xor(s, off, 32);
            if (j == 0) scal[0] = s;
        } else if (j < 64) {
            const int l = j - 32;
            float s = qb[l] * qb[l] + qb[l + 32] * qb[l + 32] +
                      qb[l + 64] * qb[l + 64] + qb[l + 96] * qb[l + 96];
            for (int off = 16; off; off >>= 1) s += __shfl_xor(s, off, 32);
            if (l == 0) scal[1] = s;
        }
        __syncthreads();

        const float rsk = 1.f / fmaxf(sqrtf(scal[0]), 1e-12f);
        const float rsq = 1.f / fmaxf(sqrtf(scal[1]), 1e-12f);

        float sk = 0.f;
        #pragma unroll
        for (int i = 0; i < DK; ++i) sk += kb[i] * S[i];
        const float dv = bet * (vv - rsk * sk);

        float o = 0.f;
        #pragma unroll
        for (int i = 0; i < DK; ++i) {
            float s2 = S[i] * dec + (kb[i] * rsk) * dv;
            S[i] = s2;
            o += qb[i] * s2;
        }
        o *= rsq;

        o2[j] = o * o;
        __syncthreads();
        if (j < 32) {
            float s = o2[j] + o2[j + 32] + o2[j + 64] + o2[j + 96];
            for (int off = 16; off; off >>= 1) s += __shfl_xor(s, off, 32);
            if (j == 0) scal[2] = s;
        } else if (j < 64) {
            const int l = j - 32;
            float s = o2[128 + l] + o2[128 + l + 32] + o2[128 + l + 64] + o2[128 + l + 96];
            for (int off = 16; off; off >>= 1) s += __shfl_xor(s, off, 32);
            if (l == 0) scal[3] = s;
        }
        __syncthreads();

        const float rms = rsqrtf(scal[2 + p] * (1.f / (float)DV) + 1e-6f);
        const float sig = 1.f / (1.f + expf(-zv));
        const float g = o * rms * nw * (zv * sig);
        const size_t oi = ((size_t)b * TT + t) * VD + h * SV + j;
        const unsigned short gh = f2bf(g);
        og_hi[oi] = gh;
        og_lo[oi] = f2bf(g - bf2f(gh));
    }

    // final state: S_final[b, h, i, p*DV + jv]
    const size_t so = (size_t)bh * DK * SV;
    #pragma unroll
    for (int i = 0; i < DK; ++i)
        S_out[so + (size_t)i * SV + j] = S[i];
}

// =====================================================================
extern "C" void kernel_launch(void* const* d_in, const int* in_sizes, int n_in,
                              void* d_out, int out_size, void* d_ws, size_t ws_size,
                              hipStream_t stream) {
    const float* x       = (const float*)d_in[0];
    const float* W_qkv   = (const float*)d_in[1];
    const float* W_z     = (const float*)d_in[2];
    const float* W_b     = (const float*)d_in[3];
    const float* W_a     = (const float*)d_in[4];
    const float* conv_w  = (const float*)d_in[5];
    const float* dt_bias = (const float*)d_in[6];
    const float* A_log   = (const float*)d_in[7];
    const float* norm_w  = (const float*)d_in[8];
    const float* W_out   = (const float*)d_in[9];

    float* ws    = (float*)d_ws;
    float* qkv   = ws;                         // 33,554,432 f32 (recycled below)
    float* qkvc  = ws + 33554432;              // 33,554,432 f32
    float* zbuf  = ws + 67108864;              // 16,777,216 f32
    float* betab = ws + 83886080;              // 131,072
    float* decb  = betab + 131072;             // 131,072
    unsigned short* bfpool = (unsigned short*)(ws + 84148224);
    unsigned short* xhi    = bfpool;                         //  8,388,608 ush
    unsigned short* xlo    = xhi + (size_t)MROWS * HID;
    unsigned short* wqh    = xlo + (size_t)MROWS * HID;      // 16,777,216 ush
    unsigned short* wql    = wqh + (size_t)QKV * HID;
    unsigned short* wzh    = wql + (size_t)QKV * HID;        //  8,388,608 ush
    unsigned short* wzl    = wzh + (size_t)VD * HID;
    // recycled qkv region (after conv) hosts og + W_out bf16 planes:
    unsigned short* ogh    = (unsigned short*)qkv;           // 16,777,216 ush
    unsigned short* ogl    = ogh + (size_t)MROWS * VD;
    unsigned short* woh    = ogl + (size_t)MROWS * VD;       //  8,388,608 ush
    unsigned short* wol    = woh + (size_t)HID * VD;

    float* y     = (float*)d_out;              // [B,T,HID]
    float* S_out = y + (size_t)BB * TT * HID;  // [B,NK,DK,SV]
    float* cbuf  = S_out + (size_t)BB * NK * DK * SV; // [B,QKV,KC-1]

    // 0) split f32 -> bf16 hi/lo planes (bandwidth-bound, one pass)
    cvt_kernel<<<(int)(((size_t)MROWS * HID / 4 + 255) / 256), 256, 0, stream>>>(x, xhi, xlo, (long long)MROWS * HID / 4);
    cvt_kernel<<<(int)(((size_t)QKV * HID / 4 + 255) / 256), 256, 0, stream>>>(W_qkv, wqh, wql, (long long)QKV * HID / 4);
    cvt_kernel<<<(int)(((size_t)VD * HID / 4 + 255) / 256), 256, 0, stream>>>(W_z, wzh, wzl, (long long)VD * HID / 4);

    // 1) qkv = x @ W_qkv^T     M=4096 N=8192 K=2048
    gemm_tdm<<<dim3(QKV / 64, MROWS / 128), 256, 0, stream>>>(xhi, xlo, wqh, wql, qkv, MROWS, QKV, HID);
    // 2) z = x @ W_z^T         M=4096 N=4096 K=2048
    gemm_tdm<<<dim3(VD / 64, MROWS / 128), 256, 0, stream>>>(xhi, xlo, wzh, wzl, zbuf, MROWS, VD, HID);
    // 3) beta / decay
    betadt_kernel<<<MROWS, 64, 0, stream>>>(x, W_b, W_a, dt_bias, A_log, betab, decb);
    // 4) causal depthwise conv + conv buffer output
    conv_kernel<<<(int)(((size_t)BB * TT * QKV) / 256), 256, 0, stream>>>(qkv, conv_w, qkvc);
    convbuf_kernel<<<(BB * QKV * (KC - 1) + 255) / 256, 256, 0, stream>>>(qkv, cbuf);
    // 5) qkv region now dead -> convert W_out into it
    cvt_kernel<<<(int)(((size_t)HID * VD / 4 + 255) / 256), 256, 0, stream>>>(W_out, woh, wol, (long long)HID * VD / 4);
    // 6) gated delta-rule scan (RMSNorm + silu(z) gate), writes og bf16 hi/lo
    scan_kernel<<<BB * NK, 256, 0, stream>>>(qkvc, zbuf, betab, decb, norm_w, ogh, ogl, S_out);
    // 7) y = og @ W_out^T      M=4096 N=2048 K=4096
    gemm_tdm<<<dim3(HID / 64, MROWS / 128), 256, 0, stream>>>(ogh, ogl, woh, wol, y, MROWS, HID, VD);
}
